// Recurrence_33509334844129
// MI455X (gfx1250) — matrix operand order repaired and checked by reading, not compiled, not run on hardware
//
#include <hip/hip_runtime.h>
#include <hip/hip_bf16.h>
#include <cstddef>

typedef __attribute__((ext_vector_type(16))) _Float16 v16h;
typedef __attribute__((ext_vector_type(8)))  _Float16 v8h;
typedef __attribute__((ext_vector_type(8)))  float    v8f;

static constexpr int kT   = 64;
static constexpr int kN   = 256;
static constexpr int kNL  = 16;
static constexpr int kENC = 512;
static constexpr int kHID = 1024;
static constexpr int kNE  = 4;
static constexpr int kGIN = 1537;           // 1 + 512 + 1024
static constexpr int kEB  = kNL * kN;       // 4096 encoder batch
static constexpr int kOW  = 1124;           // per-(t,n) output width

__device__ __forceinline__ float sigmf(float x) { return 1.0f / (1.0f + __expf(-x)); }

// ======================= WMMA GEMM: C = act(A @ B^T + bias) =======================
// A: (M,K) f16 row-major. B: (Nout,K) f16 row-major (weight rows = output cols).
// Wave tile (16*MT)x64: MT A-fragments share 4 B-fragments -> 4*MT WMMAs/K-step.
// Block = 4 waves. Requires K%32==0, Nout%64==0, M%(16*MT)==0.
template <int MT, int ACT, bool S32, bool S16>
__global__ void wmma_gemm(const _Float16* __restrict__ A,
                          const _Float16* __restrict__ B,
                          const float* __restrict__ bias,
                          float* __restrict__ C32,
                          _Float16* __restrict__ C16,
                          int M, int Nout, int K)
{
  const int lane = threadIdx.x & 31;
  const int wav  = threadIdx.x >> 5;
  const int half = lane >> 4;      // which half-wave
  const int r    = lane & 15;

  const int m0 = blockIdx.x * (64 * MT) + wav * (16 * MT);
  const int n0 = blockIdx.y * 64;
  if (m0 >= M) return;

  v8f acc[MT][4] = {};

  // A fragment (ISA 16-bit A 16x32 layout): lanes<16 hold K {0..7,16..23},
  // lanes>=16 hold K {8..15,24..31} of row M=lane&15 -> two contiguous v8h loads.
  const _Float16* arow[MT];
#pragma unroll
  for (int mt = 0; mt < MT; ++mt)
    arow[mt] = A + (size_t)(m0 + mt * 16 + r) * K + half * 8;

  for (int k0 = 0; k0 < K; k0 += 32) {
    v16h a[MT];
#pragma unroll
    for (int mt = 0; mt < MT; ++mt) {
      v8h alo = *(const v8h*)(arow[mt] + k0);
      v8h ahi = *(const v8h*)(arow[mt] + k0 + 16);
#pragma unroll
      for (int i = 0; i < 8; ++i) { a[mt][i] = alo[i]; a[mt][8 + i] = ahi[i]; }
    }
#pragma unroll
    for (int t = 0; t < 4; ++t) {
      // B fragment (32x16): lane holds column n0+t*16+(lane&15), K {0..7,16..23} / {8..15,24..31}
      const _Float16* bptr = B + (size_t)(n0 + t * 16 + r) * K + k0 + half * 8;
      v8h blo = *(const v8h*)bptr;
      v8h bhi = *(const v8h*)(bptr + 16);
      v16h b;
#pragma unroll
      for (int i = 0; i < 8; ++i) { b[i] = blo[i]; b[8 + i] = bhi[i]; }
#pragma unroll
      for (int mt = 0; mt < MT; ++mt)
        acc[mt][t] = __builtin_amdgcn_wmma_f32_16x16x32_f16(
            false, a[mt], false, b, (short)0, acc[mt][t], false, false);
    }
  }

  // C layout: VGPR rr -> row base+rr+8*half; lane&15 -> column.
#pragma unroll
  for (int mt = 0; mt < MT; ++mt) {
#pragma unroll
    for (int t = 0; t < 4; ++t) {
      const int ncol = n0 + t * 16 + r;
      const float bv = bias ? bias[ncol] : 0.0f;
#pragma unroll
      for (int rr = 0; rr < 8; ++rr) {
        const int mrow = m0 + mt * 16 + rr + 8 * half;
        float v = acc[mt][t][rr] + bv;
        if (ACT == 1) v = v > 0.0f ? v : 0.0f;
        const size_t idx = (size_t)mrow * Nout + ncol;
        if (S32) C32[idx] = v;
        if (S16) C16[idx] = (_Float16)v;
      }
    }
  }
}

// ======================= small utility kernels =======================
__global__ void k_f32_to_f16(const float* __restrict__ s, _Float16* __restrict__ d, long n) {
  long i = (long)blockIdx.x * blockDim.x + threadIdx.x;
  if (i < n) d[i] = (_Float16)s[i];
}

__global__ void k_strided_f16(const float* __restrict__ s, _Float16* __restrict__ d,
                              int rows, int cols, int ld, int off) {
  long i = (long)blockIdx.x * blockDim.x + threadIdx.x;
  long n = (long)rows * cols;
  if (i < n) {
    int rr = (int)(i / cols), c = (int)(i % cols);
    d[i] = (_Float16)s[(long)rr * ld + off + c];
  }
}

// M[n*16+l,:] = embed_task[lines[n*16+l],:]  (f32 + f16 copies)
__global__ void k_gather_M(const float* __restrict__ et, const int* __restrict__ lines,
                           float* __restrict__ M32, _Float16* __restrict__ M16) {
  long i = (long)blockIdx.x * blockDim.x + threadIdx.x;
  if (i < (long)kEB * kENC) {
    int row = (int)(i >> 9), c = (int)(i & 511);
    float v = et[(long)lines[row] * kENC + c];
    M32[i] = v; M16[i] = (_Float16)v;
  }
}

// ======================= encoder GRU gate =======================
// batch bi = i*kN + n ; input at step = MX[n*16 + (i+jeff)%16]
__global__ void k_enc_gate(const float* __restrict__ MX, const float* __restrict__ gh,
                           float* __restrict__ h, _Float16* __restrict__ h16, int jeff) {
  long idx = (long)blockIdx.x * blockDim.x + threadIdx.x;
  if (idx >= (long)kEB * kENC) return;
  int bi = (int)(idx >> 9), c = (int)(idx & 511);
  int n = bi & (kN - 1), i = bi >> 8;
  int l = (i + jeff) & (kNL - 1);
  const float* gir = MX + (size_t)(n * kNL + l) * (3 * kENC);
  const float* ghr = gh + (size_t)bi * (3 * kENC);
  float r  = sigmf(gir[c]            + ghr[c]);
  float z  = sigmf(gir[kENC + c]     + ghr[kENC + c]);
  float nn = tanhf(gir[2 * kENC + c] + r * ghr[2 * kENC + c]);
  float hv = (1.0f - z) * nn + z * h[idx];
  h[idx] = hv; h16[idx] = (_Float16)hv;
}

// B0[bi, j, dir, e] = sigmoid(h1[bi,:] . beta_W2[e,:] + b2[e])
__global__ void k_beta2(const _Float16* __restrict__ h1, const float* __restrict__ W2,
                        const float* __restrict__ b2, float* __restrict__ B0, int j, int dir) {
  int tid = blockIdx.x * blockDim.x + threadIdx.x;
  if (tid >= kEB * kNE) return;
  int bi = tid >> 2, e = tid & 3;
  const _Float16* hr = h1 + (size_t)bi * kENC;
  const float* wr = W2 + (size_t)e * kENC;
  float s = b2[e];
  for (int q = 0; q < kENC; ++q) s += (float)hr[q] * wr[q];
  B0[(size_t)bi * 128 + j * 8 + dir * 4 + e] = sigmf(s);
}

// masks + cumprod + reorder -> Pn (N, NL, 32, NE)
__global__ void k_compute_P(const float* __restrict__ B0, float* __restrict__ Pn) {
  int tid = blockIdx.x * blockDim.x + threadIdx.x;
  if (tid >= kEB) return;
  int i = tid / kN, n = tid % kN;
  const float* base = B0 + (size_t)(i * kN + n) * 128;
  for (int e = 0; e < kNE; ++e) {
    float zl[32];
    for (int j = 0; j < kNL; ++j) {
      zl[2 * j]     = base[(j * 2 + 0) * 4 + e];              // f[j]
      zl[2 * j + 1] = base[((kNL - 1 - j) * 2 + 1) * 4 + e];  // b[15-j]
    }
    zl[0] = 0.0f; zl[31] = 0.0f;          // mfirst / mlast zeroing
    float P5[32]; float c = 1.0f;
    for (int k = 0; k < 32; ++k) {
      float bf = (k == 31) ? 1.0f : zl[k];
      P5[k] = bf * c;
      c *= (1.0f - zl[k]);
    }
    for (int k = 0; k < 32; ++k) {
      float pv = (k < kNL) ? P5[2 * (kNL - 1 - k) + 1] : P5[2 * (k - kNL)];
      Pn[(((size_t)n * kNL + i) * 32 + k) * kNE + e] = pv;
    }
  }
}

// ======================= time loop kernels =======================
__global__ void k_assemble_gi(const float* __restrict__ obs, const int* __restrict__ act_a,
                              const float* __restrict__ MQ, const float* __restrict__ EA,
                              const float* __restrict__ gru_Wih, const int* __restrict__ pbuf,
                              float* __restrict__ gi, int t) {
  long idx = (long)blockIdx.x * blockDim.x + threadIdx.x;
  if (idx >= (long)kN * 3 * kHID) return;
  int n = (int)(idx / (3 * kHID)), q = (int)(idx % (3 * kHID));
  int pa = (t == 0) ? 0 : act_a[(t - 1) * kN + n];
  float o = obs[t * kN + n];
  gi[idx] = o * gru_Wih[(size_t)q * kGIN]
          + MQ[(size_t)(n * kNL + pbuf[n]) * (3 * kHID) + q]
          + EA[(size_t)pa * (3 * kHID) + q];   // EA has gru_bih folded in
}

__global__ void k_gru_gate_t(const float* __restrict__ gi, const float* __restrict__ gh,
                             float* __restrict__ h, _Float16* __restrict__ h16,
                             float* __restrict__ out, int t) {
  long idx = (long)blockIdx.x * blockDim.x + threadIdx.x;
  if (idx >= (long)kN * kHID) return;
  int n = (int)(idx / kHID), c = (int)(idx % kHID);
  const float* gir = gi + (size_t)n * (3 * kHID);
  const float* ghr = gh + (size_t)n * (3 * kHID);
  float r  = sigmf(gir[c]            + ghr[c]);
  float z  = sigmf(gir[kHID + c]     + ghr[kHID + c]);
  float nn = tanhf(gir[2 * kHID + c] + r * ghr[2 * kHID + c]);
  float hv = (1.0f - z) * nn + z * h[idx];
  h[idx] = hv; h16[idx] = (_Float16)hv;
  out[((size_t)t * kN + n) * kOW + 4 + c] = hv;
}

// one wave per batch row n: actor/ups/critic heads, softmaxes, d_probs, p update
__global__ void k_heads(const float* __restrict__ z,
                        const float* __restrict__ aW, const float* __restrict__ ab,
                        const float* __restrict__ uW, const float* __restrict__ ub,
                        const float* __restrict__ cW, const float* __restrict__ cb,
                        const float* __restrict__ Pn, int* __restrict__ pbuf,
                        const int* __restrict__ act_a, const int* __restrict__ act_d,
                        float* __restrict__ out, int t) {
  const int lane = threadIdx.x & 31;
  const int wav  = threadIdx.x >> 5;
  const int n = blockIdx.x * (blockDim.x >> 5) + wav;
  if (n >= kN) return;

  const float* zr = z + (size_t)n * kHID;

  float la0 = ab[lane], la1 = ab[lane + 32];
  {
    const float* w0 = aW + (size_t)lane * kHID;
    const float* w1 = aW + (size_t)(lane + 32) * kHID;
    for (int q = 0; q < kHID; ++q) { float zv = zr[q]; la0 += zv * w0[q]; la1 += zv * w1[q]; }
  }
  float extra = 0.0f;
  {
    const float* ew = nullptr; float eb = 0.0f;
    if (lane < 4)       { ew = uW + (size_t)lane * kHID; eb = ub[lane]; }
    else if (lane == 4) { ew = cW;                       eb = cb[0]; }
    if (ew) { float s = eb; for (int q = 0; q < kHID; ++q) s += zr[q] * ew[q]; extra = s; }
  }

  // softmax over 64 actor logits (2 per lane)
  float mx = fmaxf(la0, la1);
  for (int o = 16; o > 0; o >>= 1) mx = fmaxf(mx, __shfl_xor(mx, o, 32));
  float e0 = __expf(la0 - mx), e1 = __expf(la1 - mx);
  float sm = e0 + e1;
  for (int o = 16; o > 0; o >>= 1) sm += __shfl_xor(sm, o, 32);
  float inv = 1.0f / sm;

  // softmax over 4 ups logits; critic from lane 4
  float u0 = __shfl(extra, 0, 32), u1 = __shfl(extra, 1, 32);
  float u2 = __shfl(extra, 2, 32), u3 = __shfl(extra, 3, 32);
  float vv = __shfl(extra, 4, 32);
  float um = fmaxf(fmaxf(u0, u1), fmaxf(u2, u3));
  float q0 = __expf(u0 - um), q1 = __expf(u1 - um), q2 = __expf(u2 - um), q3 = __expf(u3 - um);
  float ui = 1.0f / (q0 + q1 + q2 + q3);
  q0 *= ui; q1 *= ui; q2 *= ui; q3 *= ui;

  const int p = pbuf[n];
  const float* wrow = Pn + (((size_t)n * kNL + p) * 32 + lane) * kNE;
  float dp = wrow[0] * q0 + wrow[1] * q1 + wrow[2] * q2 + wrow[3] * q3;

  float* ob = out + ((size_t)t * kN + n) * kOW;
  ob[1028 + lane]      = e0 * inv;
  ob[1028 + lane + 32] = e1 * inv;
  ob[1092 + lane]      = dp;

  if (lane == 0) {
    const int at = act_a[t * kN + n];
    const int dt = act_d[t * kN + n];
    int pn = p + dt - kNL;
    pn = pn < 0 ? 0 : (pn > kNL - 1 ? kNL - 1 : pn);
    ob[0] = (float)at; ob[1] = (float)dt; ob[2] = (float)pn; ob[3] = vv;
    pbuf[n] = pn;
  }
}

// ======================= host-side helpers =======================
static inline size_t align_up(size_t x, size_t a) { return (x + a - 1) & ~(a - 1); }
static inline dim3 g1(long n, int b) { return dim3((unsigned)((n + b - 1) / b)); }

// Dispatch: big-M GEMMs use 32-row wave tiles (2x arithmetic intensity on B),
// small-M (latency-bound sequential loop) keeps 16-row tiles for more blocks.
static inline void gemmA(const _Float16* A, const _Float16* B, const float* bias,
                         float* C32, int M, int N, int K, hipStream_t st) {
  dim3 b(128);
  if (M >= 1024) {
    dim3 g((M + 127) / 128, N / 64);
    wmma_gemm<2, 0, true, false><<<g, b, 0, st>>>(A, B, bias, C32, nullptr, M, N, K);
  } else {
    dim3 g((M + 63) / 64, N / 64);
    wmma_gemm<1, 0, true, false><<<g, b, 0, st>>>(A, B, bias, C32, nullptr, M, N, K);
  }
}
static inline void gemmR16(const _Float16* A, const _Float16* B, const float* bias,
                           _Float16* C16, int M, int N, int K, hipStream_t st) {
  dim3 b(128);
  if (M >= 1024) {
    dim3 g((M + 127) / 128, N / 64);
    wmma_gemm<2, 1, false, true><<<g, b, 0, st>>>(A, B, bias, nullptr, C16, M, N, K);
  } else {
    dim3 g((M + 63) / 64, N / 64);
    wmma_gemm<1, 1, false, true><<<g, b, 0, st>>>(A, B, bias, nullptr, C16, M, N, K);
  }
}
static inline void gemmR32(const _Float16* A, const _Float16* B, const float* bias,
                           float* C32, int M, int N, int K, hipStream_t st) {
  dim3 b(128);
  dim3 g((M + 63) / 64, N / 64);
  wmma_gemm<1, 1, true, false><<<g, b, 0, st>>>(A, B, bias, C32, nullptr, M, N, K);
}

extern "C" void kernel_launch(void* const* d_in, const int* in_sizes, int n_in,
                              void* d_out, int out_size, void* d_ws, size_t ws_size,
                              hipStream_t stream) {
  const float* obs        = (const float*)d_in[0];
  const int*   lines      = (const int*)  d_in[1];
  const int*   actions_a  = (const int*)  d_in[2];
  const int*   actions_d  = (const int*)  d_in[3];
  const float* embed_task = (const float*)d_in[4];
  const float* embed_act  = (const float*)d_in[5];
  const float* te_f_Wih   = (const float*)d_in[6];
  const float* te_f_Whh   = (const float*)d_in[7];
  const float* te_f_bih   = (const float*)d_in[8];
  const float* te_f_bhh   = (const float*)d_in[9];
  const float* te_b_Wih   = (const float*)d_in[10];
  const float* te_b_Whh   = (const float*)d_in[11];
  const float* te_b_bih   = (const float*)d_in[12];
  const float* te_b_bhh   = (const float*)d_in[13];
  const float* beta_W1    = (const float*)d_in[14];
  const float* beta_b1    = (const float*)d_in[15];
  const float* beta_W2    = (const float*)d_in[16];
  const float* beta_b2    = (const float*)d_in[17];
  const float* gru_Wih    = (const float*)d_in[18];
  const float* gru_Whh    = (const float*)d_in[19];
  const float* gru_bih    = (const float*)d_in[20];
  const float* gru_bhh    = (const float*)d_in[21];
  const float* zeta_W1    = (const float*)d_in[22];
  const float* zeta_b1    = (const float*)d_in[23];
  const float* zeta_W2    = (const float*)d_in[24];
  const float* zeta_b2    = (const float*)d_in[25];
  const float* ups_W      = (const float*)d_in[26];
  const float* ups_b      = (const float*)d_in[27];
  const float* actor_W    = (const float*)d_in[28];
  const float* actor_b    = (const float*)d_in[29];
  const float* critic_W   = (const float*)d_in[30];
  const float* critic_b   = (const float*)d_in[31];
  float* out = (float*)d_out;

  // ---- workspace bump allocator (~196 MB total) ----
  char* ws = (char*)d_ws; size_t off = 0;
  auto alloc = [&](size_t bytes) -> void* {
    void* p = ws + off; off = align_up(off + bytes, 256); return p;
  };
  float*    M32      = (float*)   alloc((size_t)kEB * kENC * 4);
  _Float16* M16      = (_Float16*)alloc((size_t)kEB * kENC * 2);
  _Float16* fWih16   = (_Float16*)alloc((size_t)3 * kENC * kENC * 2);
  _Float16* fWhh16   = (_Float16*)alloc((size_t)3 * kENC * kENC * 2);
  _Float16* bWih16   = (_Float16*)alloc((size_t)3 * kENC * kENC * 2);
  _Float16* bWhh16   = (_Float16*)alloc((size_t)3 * kENC * kENC * 2);
  _Float16* betaW116 = (_Float16*)alloc((size_t)kENC * kENC * 2);
  _Float16* gWhh16   = (_Float16*)alloc((size_t)3 * kHID * kHID * 2);
  _Float16* WihM16   = (_Float16*)alloc((size_t)3 * kHID * kENC * 2);
  _Float16* WihA16   = (_Float16*)alloc((size_t)3 * kHID * kHID * 2);
  _Float16* zW116    = (_Float16*)alloc((size_t)kHID * kHID * 2);
  _Float16* zW216    = (_Float16*)alloc((size_t)kHID * kHID * 2);
  _Float16* embA16   = (_Float16*)alloc((size_t)64 * kHID * 2);
  float*    MXf      = (float*)   alloc((size_t)kEB * 3 * kENC * 4);
  float*    MXb      = (float*)   alloc((size_t)kEB * 3 * kENC * 4);
  float*    MQ       = (float*)   alloc((size_t)kEB * 3 * kHID * 4);
  float*    EA       = (float*)   alloc((size_t)64 * 3 * kHID * 4);
  float*    henc     = (float*)   alloc((size_t)kEB * kENC * 4);
  _Float16* henc16   = (_Float16*)alloc((size_t)kEB * kENC * 2);
  float*    ghenc    = (float*)   alloc((size_t)kEB * 3 * kENC * 4);
  _Float16* h116     = (_Float16*)alloc((size_t)kEB * kENC * 2);
  float*    B0       = (float*)   alloc((size_t)kEB * 128 * 4);
  float*    Pn       = (float*)   alloc((size_t)kN * kNL * 32 * kNE * 4);
  float*    ht       = (float*)   alloc((size_t)kN * kHID * 4);
  _Float16* ht16     = (_Float16*)alloc((size_t)kN * kHID * 2);
  float*    gi       = (float*)   alloc((size_t)kN * 3 * kHID * 4);
  float*    ght      = (float*)   alloc((size_t)kN * 3 * kHID * 4);
  _Float16* z116     = (_Float16*)alloc((size_t)kN * kHID * 2);
  float*    z32      = (float*)   alloc((size_t)kN * kHID * 4);
  int*      pbuf     = (int*)     alloc((size_t)kN * 4);
  (void)ws_size; (void)in_sizes; (void)n_in; (void)out_size;

  const int TB = 256;

  // ---- one-time conversions / gathers ----
  k_gather_M<<<g1((long)kEB * kENC, TB), TB, 0, stream>>>(embed_task, lines, M32, M16);
  k_f32_to_f16<<<g1((long)3 * kENC * kENC, TB), TB, 0, stream>>>(te_f_Wih, fWih16, (long)3 * kENC * kENC);
  k_f32_to_f16<<<g1((long)3 * kENC * kENC, TB), TB, 0, stream>>>(te_f_Whh, fWhh16, (long)3 * kENC * kENC);
  k_f32_to_f16<<<g1((long)3 * kENC * kENC, TB), TB, 0, stream>>>(te_b_Wih, bWih16, (long)3 * kENC * kENC);
  k_f32_to_f16<<<g1((long)3 * kENC * kENC, TB), TB, 0, stream>>>(te_b_Whh, bWhh16, (long)3 * kENC * kENC);
  k_f32_to_f16<<<g1((long)kENC * kENC, TB), TB, 0, stream>>>(beta_W1, betaW116, (long)kENC * kENC);
  k_f32_to_f16<<<g1((long)3 * kHID * kHID, TB), TB, 0, stream>>>(gru_Whh, gWhh16, (long)3 * kHID * kHID);
  k_f32_to_f16<<<g1((long)kHID * kHID, TB), TB, 0, stream>>>(zeta_W1, zW116, (long)kHID * kHID);
  k_f32_to_f16<<<g1((long)kHID * kHID, TB), TB, 0, stream>>>(zeta_W2, zW216, (long)kHID * kHID);
  k_f32_to_f16<<<g1((long)64 * kHID, TB), TB, 0, stream>>>(embed_act, embA16, (long)64 * kHID);
  k_strided_f16<<<g1((long)3 * kHID * kENC, TB), TB, 0, stream>>>(gru_Wih, WihM16, 3 * kHID, kENC, kGIN, 1);
  k_strided_f16<<<g1((long)3 * kHID * kHID, TB), TB, 0, stream>>>(gru_Wih, WihA16, 3 * kHID, kHID, kGIN, 1 + kENC);

  // ---- precompute input projections (WMMA GEMMs) ----
  gemmA(M16, fWih16, te_f_bih, MXf, kEB, 3 * kENC, kENC, stream);
  gemmA(M16, bWih16, te_b_bih, MXb, kEB, 3 * kENC, kENC, stream);
  gemmA(M16, WihM16, nullptr, MQ, kEB, 3 * kHID, kENC, stream);
  gemmA(embA16, WihA16, gru_bih, EA, 64, 3 * kHID, kHID, stream);

  // ---- task encoder: forward GRU, fused beta head ----
  hipMemsetAsync(henc, 0, (size_t)kEB * kENC * 4, stream);
  hipMemsetAsync(henc16, 0, (size_t)kEB * kENC * 2, stream);
  for (int j = 0; j < kNL; ++j) {
    gemmA(henc16, fWhh16, te_f_bhh, ghenc, kEB, 3 * kENC, kENC, stream);
    k_enc_gate<<<g1((long)kEB * kENC, TB), TB, 0, stream>>>(MXf, ghenc, henc, henc16, j);
    gemmR16(henc16, betaW116, beta_b1, h116, kEB, kENC, kENC, stream);
    k_beta2<<<g1((long)kEB * kNE, TB), TB, 0, stream>>>(h116, beta_W2, beta_b2, B0, j, 0);
  }
  // ---- backward GRU ----
  hipMemsetAsync(henc, 0, (size_t)kEB * kENC * 4, stream);
  hipMemsetAsync(henc16, 0, (size_t)kEB * kENC * 2, stream);
  for (int k = 0; k < kNL; ++k) {
    gemmA(henc16, bWhh16, te_b_bhh, ghenc, kEB, 3 * kENC, kENC, stream);
    k_enc_gate<<<g1((long)kEB * kENC, TB), TB, 0, stream>>>(MXb, ghenc, henc, henc16, kNL - 1 - k);
    gemmR16(henc16, betaW116, beta_b1, h116, kEB, kENC, kENC, stream);
    k_beta2<<<g1((long)kEB * kNE, TB), TB, 0, stream>>>(h116, beta_W2, beta_b2, B0, kNL - 1 - k, 1);
  }
  k_compute_P<<<g1(kEB, TB), TB, 0, stream>>>(B0, Pn);

  // ---- time recurrence ----
  hipMemsetAsync(ht, 0, (size_t)kN * kHID * 4, stream);
  hipMemsetAsync(ht16, 0, (size_t)kN * kHID * 2, stream);
  hipMemsetAsync(pbuf, 0, (size_t)kN * 4, stream);
  for (int t = 0; t < kT; ++t) {
    k_assemble_gi<<<g1((long)kN * 3 * kHID, TB), TB, 0, stream>>>(
        obs, actions_a, MQ, EA, gru_Wih, pbuf, gi, t);
    gemmA(ht16, gWhh16, gru_bhh, ght, kN, 3 * kHID, kHID, stream);
    k_gru_gate_t<<<g1((long)kN * kHID, TB), TB, 0, stream>>>(gi, ght, ht, ht16, out, t);
    gemmR16(ht16, zW116, zeta_b1, z116, kN, kHID, kHID, stream);
    gemmR32(z116, zW216, zeta_b2, z32, kN, kHID, kHID, stream);
    k_heads<<<kN / 8, 256, 0, stream>>>(z32, actor_W, actor_b, ups_W, ups_b,
                                        critic_W, critic_b, Pn, pbuf,
                                        actions_a, actions_d, out, t);
  }
}
